// MultiAxisAttention_62362925138373
// MI455X (gfx1250) — matrix operand order repaired and checked by reading, not compiled, not run on hardware
//
#include <hip/hip_runtime.h>
#include <hip/hip_bf16.h>

typedef __bf16 bf16;
typedef __attribute__((ext_vector_type(16))) __bf16 v16bf;
typedef __attribute__((ext_vector_type(8)))  float   v8f;
typedef __attribute__((ext_vector_type(4)))  unsigned int v4u;
typedef __attribute__((ext_vector_type(4)))  float   f4v;

#define I49   49
#define IPAD  64
#define DMODEL 256
#define NHEADS 8
#define DHEAD 32
#define NWIN  4096   // 16 * 256

// ---------------- fragment loaders (CDNA5 WMMA bf16 16x16x32 layouts) ----------------
// A-matrix 16x32 (MxK): lane<16 -> row=lane, K chunks {0..7},{16..23};
//                       lane>=16 -> row=lane-16, K chunks {8..15},{24..31}
__device__ inline v16bf load_frag_a(const bf16* base, int rowStride, int lane) {
    int row = lane & 15;
    int kb  = (lane >> 4) << 3;            // 0 or 8
    const bf16* p = base + row * rowStride + kb;
    union { v16bf v; v4u u[2]; } r;
    r.u[0] = *(const v4u*)(p);             // K = kb .. kb+7   (16 B)
    r.u[1] = *(const v4u*)(p + 16);        // K = kb+16 .. kb+23
    return r.v;
}

// B-matrix 32x16 (KxN): lane<16 -> col=lane, K chunks {0..7},{16..23};
//                       lane>=16 -> col=lane-16, K chunks {8..15},{24..31}
__device__ inline v16bf load_frag_b(const bf16* base, int rowStride, int lane) {
    int col = lane & 15;
    int kb  = (lane >> 4) << 3;            // 0 or 8
    const bf16* p = base + col * rowStride + kb;
    union { v16bf v; v4u u[2]; } r;
    r.u[0] = *(const v4u*)(p);
    r.u[1] = *(const v4u*)(p + 16);
    return r.v;
}

__device__ inline v8f wmma_bf16(v16bf a, v16bf b, v8f c) {
    return __builtin_amdgcn_wmma_f32_16x16x32_bf16(false, a, false, b, (short)0, c,
                                                   false, false);
}

// ---------------- prep: weights fp32->bf16, gather rel-pos bias ----------------
__global__ void prep_kernel(const float* __restrict__ wqkv,
                            const float* __restrict__ btab,
                            const float* __restrict__ wout,
                            const int*   __restrict__ indices,
                            bf16* __restrict__ wqkv_bf,
                            bf16* __restrict__ wout_bf,
                            float* __restrict__ biasm) {
    int i = blockIdx.x * 256 + threadIdx.x;
    if (i < 3 * DMODEL * DMODEL) wqkv_bf[i] = (bf16)wqkv[i];
    if (i < DMODEL * DMODEL)     wout_bf[i] = (bf16)wout[i];
    if (i < NHEADS * I49 * I49) {
        int h  = i / (I49 * I49);
        int ij = i % (I49 * I49);
        biasm[i] = btab[indices[ij] * NHEADS + h];   // (49,49,h) -> (h,49,49)
    }
}

// ---------------- fused windowed attention: one workgroup per window ----------------
__global__ __launch_bounds__(256, 1)
void attn_kernel(const float* __restrict__ x,
                 const bf16*  __restrict__ wqkv,   // (768,256) bf16, row-major (out,in)
                 const bf16*  __restrict__ wout,   // (256,256) bf16
                 const float* __restrict__ biasm,  // (8,49,49) f32
                 float*       __restrict__ out) {
    __shared__ bf16  sXO[IPAD * DMODEL];                 // X, later reused as O (32 KB)
    __shared__ bf16  sQ [IPAD * DMODEL];                 // pre-scaled Q        (32 KB)
    __shared__ bf16  sK [IPAD * DMODEL];                 //                     (32 KB)
    __shared__ bf16  sVt[NHEADS * DHEAD * IPAD];         // V^T per head        (32 KB)
    __shared__ float sS [NHEADS * IPAD * IPAD];          // per-head scores    (128 KB)
    __shared__ float sRs[NHEADS * IPAD];                 // 1/rowsum             (2 KB)

    const int win  = blockIdx.x;
    const int tid  = threadIdx.x;
    const int lane = tid & 31;
    const int wave = tid >> 5;                           // == head id
    const float* xw = x   + (size_t)win * I49 * DMODEL;
    float*       ow = out + (size_t)win * I49 * DMODEL;

    // ---- 1) X -> bf16 LDS (non-temporal global read), zero-pad rows 49..63 ----
    #pragma unroll 4
    for (int e = tid; e < IPAD * DMODEL / 4; e += 256) {
        int row = e >> 6;                                // e / (DMODEL/4)
        f4v f = {0.f, 0.f, 0.f, 0.f};
        if (row < I49) f = __builtin_nontemporal_load(((const f4v*)xw) + e);
        bf16* d = sXO + e * 4;
        d[0] = (bf16)f.x; d[1] = (bf16)f.y; d[2] = (bf16)f.z; d[3] = (bf16)f.w;
    }
    __syncthreads();

    // ---- 2) QKV = X * Wqkv^T (M=64, N=768, K=256); N-tile outer so every weight
    //         fragment is read from L2 exactly once per workgroup ----
    const float qscale = 0.1767766952966369f;            // 32^-0.5
    const int ncol  = lane & 15;
    const int rbase = (lane < 16) ? 0 : 8;
    #pragma unroll 1
    for (int ntl = 0; ntl < 6; ++ntl) {
        int nt = wave * 6 + ntl;                         // 0..47 (16 Q, 16 K, 16 V)
        v16bf bfr[8];
        const bf16* wb = wqkv + nt * 16 * DMODEL;
        #pragma unroll
        for (int k = 0; k < 8; ++k)
            bfr[k] = load_frag_b(wb + k * 32, DMODEL, lane);

        // branchless store descriptor (uniform per wave/ntl):
        //   Q : sQ + nt*16      + ncol + m*256   (value * qscale)
        //   K : sK + (nt-16)*16 + ncol + m*256
        //   V : sVt + ((nt-32)*16 + ncol)*64 + m     (h*32+d == col)
        const bool  isQK = nt < 32;
        bf16* qkb   = (nt < 16 ? sQ + nt * 16 : sK + (nt - 16) * 16) + ncol;
        bf16* vbp   = sVt + ((nt - 32) * 16 + ncol) * IPAD;
        bf16* sbase = isQK ? qkb : vbp;
        const int   mstride = isQK ? DMODEL : 1;
        const float scl     = (nt < 16) ? qscale : 1.0f;

        #pragma unroll 1
        for (int mt = 0; mt < 4; ++mt) {
            v8f acc = {0.f,0.f,0.f,0.f,0.f,0.f,0.f,0.f};
            const bf16* abase = sXO + mt * 16 * DMODEL;
            v16bf acur = load_frag_a(abase, DMODEL, lane);      // k = 0
            #pragma unroll
            for (int k = 0; k < 8; ++k) {                       // prefetch k+1, mma k
                v16bf anext;
                if (k < 7) anext = load_frag_a(abase + (k + 1) * 32, DMODEL, lane);
                acc = wmma_bf16(acur, bfr[k], acc);
                if (k < 7) acur = anext;
            }
            #pragma unroll
            for (int r = 0; r < 8; ++r) {
                int m = mt * 16 + rbase + r;
                sbase[m * mstride] = (bf16)(acc[r] * scl);
            }
        }
    }
    __syncthreads();

    // ---- 3) per-head S = Q K^T (K=32 exact, 16 tiles) ----
    const int h = wave;
    float* Sw = sS + h * IPAD * IPAD;
    v16bf qa[4], kbf[4];
    #pragma unroll
    for (int t = 0; t < 4; ++t) {
        qa [t] = load_frag_a(sQ + t * 16 * DMODEL + h * DHEAD, DMODEL, lane);
        kbf[t] = load_frag_b(sK + t * 16 * DMODEL + h * DHEAD, DMODEL, lane);
    }
    #pragma unroll 1
    for (int mt = 0; mt < 4; ++mt) {
        #pragma unroll
        for (int jt = 0; jt < 4; ++jt) {
            v8f acc = {0.f,0.f,0.f,0.f,0.f,0.f,0.f,0.f};
            acc = wmma_bf16(qa[mt], kbf[jt], acc);
            float* dst = Sw + (jt * 16 + ncol);
            #pragma unroll
            for (int r = 0; r < 8; ++r)
                dst[(mt * 16 + rbase + r) * IPAD] = acc[r];
        }
    }

    // ---- 4) softmax (+ rel-pos bias), P written bf16 in place over S ----
    #pragma unroll 1
    for (int rr = 0; rr < 2; ++rr) {
        int i = lane + rr * 32;                          // 0..63
        bool valid = (i < I49);
        float* Srow = Sw + i * IPAD;
        const float* brow = biasm + (h * I49 + (valid ? i : 0)) * I49;
        float mx = -3.0e38f;
        for (int j = 0; j < I49; ++j) {
            float v = Srow[j] + brow[j];
            Srow[j] = v;
            mx = fmaxf(mx, v);
        }
        float sum = 0.f;
        bf16* Prow = (bf16*)Srow;
        for (int j = 0; j < I49; ++j) {                  // in-place: 2j < 4j read
            float p = valid ? __expf(Srow[j] - mx) : 0.f;
            sum += p;
            Prow[j] = (bf16)p;
        }
        #pragma unroll
        for (int j = I49; j < IPAD; ++j) Prow[j] = (bf16)0.f;
        sRs[h * IPAD + i] = valid ? (1.0f / sum) : 0.f;
    }

    // ---- 5) O = (P V) * (1/rowsum)  (M=64, N=32, K=64) -> into sXO ----
    const bf16* Pbase = (const bf16*)Sw;                 // row stride 128 bf16
    #pragma unroll 1
    for (int mt = 0; mt < 4; ++mt) {
        v16bf pa[2];
        #pragma unroll
        for (int k = 0; k < 2; ++k)
            pa[k] = load_frag_a(Pbase + mt * 16 * (2 * IPAD) + k * 32, 2 * IPAD, lane);
        #pragma unroll
        for (int nt = 0; nt < 2; ++nt) {
            v8f acc = {0.f,0.f,0.f,0.f,0.f,0.f,0.f,0.f};
            #pragma unroll
            for (int k = 0; k < 2; ++k) {
                v16bf vb = load_frag_b(sVt + h * DHEAD * IPAD + nt * 16 * IPAD + k * 32,
                                       IPAD, lane);
                acc = wmma_bf16(pa[k], vb, acc);
            }
            #pragma unroll
            for (int r = 0; r < 8; ++r) {
                int m = mt * 16 + rbase + r;
                float rs = sRs[h * IPAD + m];
                sXO[m * DMODEL + h * DHEAD + nt * 16 + ncol] = (bf16)(acc[r] * rs);
            }
        }
    }
    __syncthreads();

    // ---- 6) out = O * Wout^T (M=64, N=256, K=256); N-tile outer for weight reuse;
    //         non-temporal f32 stores, rows < 49 ----
    #pragma unroll 1
    for (int ntl = 0; ntl < 2; ++ntl) {
        int nt = wave * 2 + ntl;                         // 0..15
        v16bf bfr[8];
        const bf16* wb = wout + nt * 16 * DMODEL;
        #pragma unroll
        for (int k = 0; k < 8; ++k)
            bfr[k] = load_frag_b(wb + k * 32, DMODEL, lane);
        #pragma unroll 1
        for (int mt = 0; mt < 4; ++mt) {
            v8f acc = {0.f,0.f,0.f,0.f,0.f,0.f,0.f,0.f};
            const bf16* abase = sXO + mt * 16 * DMODEL;
            v16bf acur = load_frag_a(abase, DMODEL, lane);
            #pragma unroll
            for (int k = 0; k < 8; ++k) {
                v16bf anext;
                if (k < 7) anext = load_frag_a(abase + (k + 1) * 32, DMODEL, lane);
                acc = wmma_bf16(acur, bfr[k], acc);
                if (k < 7) acur = anext;
            }
            #pragma unroll
            for (int r = 0; r < 8; ++r) {
                int m = mt * 16 + rbase + r;
                if (m < I49)
                    __builtin_nontemporal_store(acc[r], &ow[m * DMODEL + nt * 16 + ncol]);
            }
        }
    }
}

extern "C" void kernel_launch(void* const* d_in, const int* in_sizes, int n_in,
                              void* d_out, int out_size, void* d_ws, size_t ws_size,
                              hipStream_t stream) {
    const float* x    = (const float*)d_in[0];
    const float* wqkv = (const float*)d_in[1];
    const float* btab = (const float*)d_in[2];
    const float* wout = (const float*)d_in[3];
    const int*   idx  = (const int*)d_in[4];
    float* out = (float*)d_out;

    char* ws = (char*)d_ws;
    bf16*  wqkv_bf = (bf16*)ws;                                   // 393216 B
    bf16*  wout_bf = (bf16*)(ws + 393216);                        // 131072 B
    float* biasm   = (float*)(ws + 393216 + 131072);              //  76832 B

    prep_kernel<<<(3 * DMODEL * DMODEL + 255) / 256, 256, 0, stream>>>(
        wqkv, btab, wout, idx, wqkv_bf, wout_bf, biasm);
    attn_kernel<<<NWIN, 256, 0, stream>>>(x, wqkv_bf, wout_bf, biasm, out);
}
